// MiniMaxLightningAttention_42743514530422
// MI455X (gfx1250) — matrix operand order repaired and checked by reading, not compiled, not run on hardware
//
#include <hip/hip_runtime.h>
#include <hip/hip_bf16.h>

// ---------------- problem constants ----------------
#define B_      2
#define S_      4096
#define H_      2048
#define NH_     32
#define HD_     64
#define BLOCK_  256
#define NB_     16
#define QKV_N   (3 * NH_ * HD_)   // 6144
#define M_      (B_ * S_)         // 8192

typedef __attribute__((ext_vector_type(16))) __bf16 v16bf;
typedef __attribute__((ext_vector_type(8)))  __bf16 v8bf;
typedef __attribute__((ext_vector_type(8)))  float  v8f;
typedef __attribute__((ext_vector_type(4)))  float  v4f;

// ---------------- WMMA fragment helpers (CDNA5 16-bit layouts) ----------------
// A-matrix 16x32 (bf16): lane half h -> K = {8h..8h+7} then {16+8h..16+8h+7}
__device__ __forceinline__ v16bf load_a_frag(const __bf16* row, int kk, int half) {
  v8bf lo = *(const v8bf*)(row + kk + 8 * half);
  v8bf hi = *(const v8bf*)(row + kk + 16 + 8 * half);
  v16bf r;
#pragma unroll
  for (int i = 0; i < 8; ++i) { r[i] = lo[i]; r[i + 8] = hi[i]; }
  return r;
}
// B-matrix 32x16 (bf16): lane half h -> K = {8h..8h+7} then {16+8h..16+8h+7}; row is the N-row
// (B stored [n][k] contiguous-k, i.e. the transposed operand's row)
__device__ __forceinline__ v16bf load_b_frag(const __bf16* row, int kk, int half) {
  v8bf lo = *(const v8bf*)(row + kk + 8 * half);
  v8bf hi = *(const v8bf*)(row + kk + 16 + 8 * half);
  v16bf r;
#pragma unroll
  for (int i = 0; i < 8; ++i) { r[i] = lo[i]; r[i + 8] = hi[i]; }
  return r;
}

__device__ __forceinline__ v8f wmma_bf16(v16bf a, v16bf b, v8f c) {
  return __builtin_amdgcn_wmma_f32_16x16x32_bf16(false, a, false, b, (short)0, c,
                                                 false, false);
}

// ---------------- fused GEMM: C[M,N] = act(A[M,K] @ W[N,K]^T) ----------------
// ACT: 0=none, 1=silu, 2=sigmoid. 128x128 tile, 8 waves of 32x64.
template <int ACT>
__global__ __launch_bounds__(256) void gemm_act_kernel(
    const float* __restrict__ A, const float* __restrict__ W,
    float* __restrict__ C, int M, int N, int K) {
  __shared__ __align__(16) __bf16 At[128 * 32];
  __shared__ __align__(16) __bf16 Bt[128 * 32];

  const int tid  = threadIdx.x;
  const int lane = tid & 31;
  const int wave = tid >> 5;
  const int wm   = wave & 3;          // 4 wave-rows
  const int wn   = wave >> 2;         // 2 wave-cols
  const int half = lane >> 4;
  const int l16  = lane & 15;
  const int m0   = blockIdx.y * 128;
  const int n0   = blockIdx.x * 128;

  v8f acc[2][4];
#pragma unroll
  for (int i = 0; i < 2; ++i)
#pragma unroll
    for (int j = 0; j < 4; ++j) acc[i][j] = (v8f){};

  const int lrow = tid >> 1;          // 0..127
  const int lk   = (tid & 1) * 16;    // 0 or 16

  for (int k0 = 0; k0 < K; k0 += 32) {
    __syncthreads();
    const float* ag = A + (size_t)(m0 + lrow) * K + k0 + lk;
    const float* bg = W + (size_t)(n0 + lrow) * K + k0 + lk;
    __builtin_prefetch(ag + 32, 0, 0);   // -> global_prefetch_b8
    __builtin_prefetch(bg + 32, 0, 0);
    v8bf apk, bpk;
#pragma unroll
    for (int c = 0; c < 16; c += 8) {
      v4f a0 = *(const v4f*)(ag + c);
      v4f a1 = *(const v4f*)(ag + c + 4);
      v4f b0 = *(const v4f*)(bg + c);
      v4f b1 = *(const v4f*)(bg + c + 4);
#pragma unroll
      for (int j = 0; j < 4; ++j) {
        apk[j] = (__bf16)a0[j]; apk[j + 4] = (__bf16)a1[j];
        bpk[j] = (__bf16)b0[j]; bpk[j + 4] = (__bf16)b1[j];
      }
      *(v8bf*)&At[lrow * 32 + lk + c] = apk;
      *(v8bf*)&Bt[lrow * 32 + lk + c] = bpk;
    }
    __syncthreads();

    v16bf bf[4];
#pragma unroll
    for (int nt = 0; nt < 4; ++nt)
      bf[nt] = load_b_frag(&Bt[(wn * 64 + nt * 16 + l16) * 32], 0, half);
#pragma unroll
    for (int mt = 0; mt < 2; ++mt) {
      v16bf af = load_a_frag(&At[(wm * 32 + mt * 16 + l16) * 32], 0, half);
#pragma unroll
      for (int nt = 0; nt < 4; ++nt)
        acc[mt][nt] = wmma_bf16(af, bf[nt], acc[mt][nt]);
    }
  }

  // epilogue with fused activation; C layout: M = i + 8*half, N = l16
#pragma unroll
  for (int mt = 0; mt < 2; ++mt)
#pragma unroll
    for (int nt = 0; nt < 4; ++nt)
#pragma unroll
      for (int i = 0; i < 8; ++i) {
        float v = acc[mt][nt][i];
        if (ACT == 1) v = v / (1.0f + __expf(-v));        // silu
        if (ACT == 2) v = 1.0f / (1.0f + __expf(-v));     // sigmoid
        int m = m0 + wm * 32 + mt * 16 + i + 8 * half;
        int n = n0 + wn * 64 + nt * 16 + l16;
        C[(size_t)m * N + n] = v;
      }
}

// ---------------- lightning attention block-scan ----------------
// One workgroup per (b, h). 8 waves; wave w owns q-rows [32w, 32w+32).
// KV state kept transposed (kvT[e][d]) in registers: 2 16x16 f32 tiles / wave.
__global__ __launch_bounds__(256) void lightning_attn_kernel(
    const float* __restrict__ qkv, float* __restrict__ attn,
    float* __restrict__ kv_out) {
  extern __shared__ __align__(16) char smem_raw[];
  __bf16* q_s  = (__bf16*)smem_raw;        // [256][64]   Q
  __bf16* qd_s = q_s  + 256 * 64;          // [256][64]   Q * query_decay
  __bf16* k_s  = qd_s + 256 * 64;          // [256][64]   K
  __bf16* vT_s = k_s  + 256 * 64;          // [64][256]   V transposed
  __bf16* kT_s = vT_s + 64 * 256;          // [64][256]   K^T * key_decay
  __bf16* kv_s = kT_s + 64 * 256;          // [64][64]    kvT bf16 mirror
  __bf16* s_s  = kv_s + 64 * 64;           // 8 waves x [32][32] S-tile bounce

  const int tid  = threadIdx.x;
  const int lane = tid & 31;
  const int wave = tid >> 5;
  const int half = lane >> 4;
  const int l16  = lane & 15;
  const int b    = blockIdx.x / NH_;
  const int h    = blockIdx.x % NH_;

  // slope = (2^(-8/NH))^(h+1) * factor ; LAYER_IDX = 0
  const float factor = 1.0f - 0.0f / (12.0f - 1.0f + 1e-5f) + 1e-5f;
  const float slope  = exp2f(-0.25f * (float)(h + 1)) * factor;
  const float bdecay = __expf(-slope * (float)BLOCK_);
  const float estep  = __expf(-slope);     // per-row decay ratio

  for (int i = tid; i < 64 * 64; i += 256) kv_s[i] = (__bf16)0.0f;

  v8f kvacc[2];
  kvacc[0] = (v8f){};
  kvacc[1] = (v8f){};

  __bf16* s_w = s_s + wave * 32 * 32;
  const float* qbase = qkv + (size_t)b * S_ * QKV_N + h * (3 * HD_);

  for (int blk = 0; blk < NB_; ++blk) {
    __syncthreads();
    // cooperative load: thread t stages key/query row t of this block
    {
      const float* rowp = qbase + (size_t)(blk * BLOCK_ + tid) * QKV_N;
      float kdec = __expf(-slope * (float)(BLOCK_ - (tid + 1)));
      float qdec = __expf(-slope * (float)(tid + 1));
#pragma unroll
      for (int i = 0; i < 64; i += 8) {
        v8bf q8, qd8, k8;
#pragma unroll
        for (int c = 0; c < 8; c += 4) {
          v4f qv = *(const v4f*)(rowp + i + c);
          v4f kv = *(const v4f*)(rowp + 64 + i + c);
          v4f vv = *(const v4f*)(rowp + 128 + i + c);
#pragma unroll
          for (int j = 0; j < 4; ++j) {
            q8[c + j]  = (__bf16)qv[j];
            qd8[c + j] = (__bf16)(qv[j] * qdec);
            k8[c + j]  = (__bf16)kv[j];
            vT_s[(i + c + j) * 256 + tid] = (__bf16)vv[j];
            kT_s[(i + c + j) * 256 + tid] = (__bf16)(kv[j] * kdec);
          }
        }
        *(v8bf*)&q_s[tid * 64 + i]  = q8;
        *(v8bf*)&qd_s[tid * 64 + i] = qd8;
        *(v8bf*)&k_s[tid * 64 + i]  = k8;
      }
    }
    __syncthreads();

    v8f oacc[2][4];
#pragma unroll
    for (int i = 0; i < 2; ++i)
#pragma unroll
      for (int j = 0; j < 4; ++j) oacc[i][j] = (v8f){};

    // hoisted Q fragments (plain and decay-scaled), reused across key-tiles
    v16bf qfrag[2][2], qdfrag[2][2];
#pragma unroll
    for (int mt = 0; mt < 2; ++mt)
#pragma unroll
      for (int kkI = 0; kkI < 2; ++kkI) {
        const int qrow = wave * 32 + mt * 16 + l16;
        qfrag[mt][kkI]  = load_a_frag(&q_s[qrow * 64], kkI * 32, half);
        qdfrag[mt][kkI] = load_a_frag(&qd_s[qrow * 64], kkI * 32, half);
      }

    // ---- inter-block: (Q * query_decay) @ KV (contraction over d) ----
#pragma unroll
    for (int mt = 0; mt < 2; ++mt)
#pragma unroll
      for (int kkI = 0; kkI < 2; ++kkI)
#pragma unroll
        for (int nt = 0; nt < 4; ++nt) {
          v16bf bf = load_b_frag(&kv_s[(nt * 16 + l16) * 64], kkI * 32, half);
          oacc[mt][nt] = wmma_bf16(qdfrag[mt][kkI], bf, oacc[mt][nt]);
        }

    // ---- intra-block: (QK^T * diag_decay) @ V over 8 key-tiles of 32 ----
    for (int kt = 0; kt < 8; ++kt) {
      v8f sacc[2][2];
#pragma unroll
      for (int i = 0; i < 2; ++i)
#pragma unroll
        for (int j = 0; j < 2; ++j) sacc[i][j] = (v8f){};
#pragma unroll
      for (int kkI = 0; kkI < 2; ++kkI)
#pragma unroll
        for (int mt = 0; mt < 2; ++mt)
#pragma unroll
          for (int nt = 0; nt < 2; ++nt) {
            v16bf bf = load_b_frag(&k_s[(kt * 32 + nt * 16 + l16) * 64],
                                   kkI * 32, half);
            sacc[mt][nt] = wmma_bf16(qfrag[mt][kkI], bf, sacc[mt][nt]);
          }
      // causal decay mask in C-layout via 1 exp + multiply recurrence,
      // bounce to LDS as bf16 A-operand
#pragma unroll
      for (int mt = 0; mt < 2; ++mt)
#pragma unroll
        for (int nt = 0; nt < 2; ++nt) {
          const int key    = kt * 32 + nt * 16 + l16;
          const int qbase0 = wave * 32 + mt * 16 + 8 * half;
          float d = __expf(-slope * (float)(qbase0 - key));
#pragma unroll
          for (int i = 0; i < 8; ++i) {
            float m = (qbase0 + i >= key) ? d : 0.0f;
            s_w[(mt * 16 + i + 8 * half) * 32 + nt * 16 + l16] =
                (__bf16)(sacc[mt][nt][i] * m);
            d *= estep;
          }
        }
      asm volatile("s_wait_dscnt 0" ::: "memory");
#pragma unroll
      for (int mt = 0; mt < 2; ++mt) {
        v16bf af = load_a_frag(&s_w[(mt * 16 + l16) * 32], 0, half);
#pragma unroll
        for (int nt = 0; nt < 4; ++nt) {
          v16bf bf = load_b_frag(&vT_s[(nt * 16 + l16) * 256], kt * 32, half);
          oacc[mt][nt] = wmma_bf16(af, bf, oacc[mt][nt]);
        }
      }
    }

    // store this wave's 32-row output strip (pre-norm attn, f32)
#pragma unroll
    for (int mt = 0; mt < 2; ++mt)
#pragma unroll
      for (int nt = 0; nt < 4; ++nt)
#pragma unroll
        for (int i = 0; i < 8; ++i) {
          int qrow = wave * 32 + mt * 16 + i + 8 * half;
          size_t s = (size_t)b * S_ + blk * BLOCK_ + qrow;
          attn[s * (NH_ * HD_) + h * HD_ + nt * 16 + l16] = oacc[mt][nt][i];
        }

    // ---- KV update: kvT = kvT*bdecay + V^T @ (K*key_decay) ----
#pragma unroll
    for (int t = 0; t < 2; ++t) {
      int id = wave * 2 + t;
      int et = id >> 2, dt = id & 3;
#pragma unroll
      for (int i = 0; i < 8; ++i) kvacc[t][i] *= bdecay;
      for (int kk = 0; kk < BLOCK_; kk += 32) {
        v16bf af = load_a_frag(&vT_s[(et * 16 + l16) * 256], kk, half);
        v16bf bf = load_b_frag(&kT_s[(dt * 16 + l16) * 256], kk, half);
        kvacc[t] = wmma_bf16(af, bf, kvacc[t]);
      }
    }
    __syncthreads();  // all inter-reads of kv_s done before refresh
#pragma unroll
    for (int t = 0; t < 2; ++t) {
      int id = wave * 2 + t, et = id >> 2, dt = id & 3;
#pragma unroll
      for (int i = 0; i < 8; ++i) {
        int e = et * 16 + i + 8 * half;
        int d = dt * 16 + l16;
        kv_s[e * 64 + d] = (__bf16)kvacc[t][i];
      }
    }
  }

  // kv_final[b][h][d][e] = kvT[e][d]
#pragma unroll
  for (int t = 0; t < 2; ++t) {
    int id = wave * 2 + t, et = id >> 2, dt = id & 3;
#pragma unroll
    for (int i = 0; i < 8; ++i) {
      int e = et * 16 + i + 8 * half;
      int d = dt * 16 + l16;
      kv_out[(((size_t)b * NH_ + h) * HD_ + d) * HD_ + e] = kvacc[t][i];
    }
  }
}

// ---------------- fused RMSNorm * sigmoid-gate (in place) ----------------
__global__ __launch_bounds__(256) void rmsnorm_gate_kernel(
    float* __restrict__ attn, const float* __restrict__ gate,
    const float* __restrict__ norm_w) {
  __shared__ float red[8];
  const int row = blockIdx.x;
  const int tid = threadIdx.x;
  float* a = attn + (size_t)row * H_;
  const float* g = gate + (size_t)row * H_;

  float v[8], ss = 0.0f;
#pragma unroll
  for (int i = 0; i < 8; ++i) {
    v[i] = a[tid + i * 256];
    ss += v[i] * v[i];
  }
#pragma unroll
  for (int off = 16; off > 0; off >>= 1) ss += __shfl_down(ss, off, 32);
  if ((tid & 31) == 0) red[tid >> 5] = ss;
  __syncthreads();
  float tot = 0.0f;
#pragma unroll
  for (int i = 0; i < 8; ++i) tot += red[i];
  float inv = rsqrtf(tot * (1.0f / (float)H_) + 1e-5f);
#pragma unroll
  for (int i = 0; i < 8; ++i) {
    int c = tid + i * 256;
    a[c] = norm_w[c] * v[i] * inv * g[c];  // gate pre-sigmoided by GEMM epilogue
  }
}

// ---------------- host launcher ----------------
extern "C" void kernel_launch(void* const* d_in, const int* in_sizes, int n_in,
                              void* d_out, int out_size, void* d_ws,
                              size_t ws_size, hipStream_t stream) {
  const float* hidden = (const float*)d_in[0];
  const float* qkv_w  = (const float*)d_in[1];
  const float* out_w  = (const float*)d_in[2];
  const float* gate_w = (const float*)d_in[3];
  const float* norm_w = (const float*)d_in[4];

  float* out      = (float*)d_out;
  float* kv_final = out + (size_t)M_ * H_;   // tuple tail: (B,NH,HD,HD)

  float* ws_qkv  = (float*)d_ws;                        // M_ x 6144
  float* ws_gate = ws_qkv + (size_t)M_ * QKV_N;         // M_ x 2048 (sigmoid'd)
  float* ws_attn = ws_gate + (size_t)M_ * H_;           // M_ x 2048

  dim3 blk(256);

  // 1) qkv = silu(hidden @ qkv_w^T)
  gemm_act_kernel<1><<<dim3(QKV_N / 128, M_ / 128), blk, 0, stream>>>(
      hidden, qkv_w, ws_qkv, M_, QKV_N, H_);
  // 2) gate = sigmoid(hidden @ gate_w^T)
  gemm_act_kernel<2><<<dim3(H_ / 128, M_ / 128), blk, 0, stream>>>(
      hidden, gate_w, ws_gate, M_, H_, H_);
  // 3) lightning attention block scan (one WG per (b,h)); 184 KB dynamic LDS
  size_t smem = (size_t)(3 * 256 * 64 + 2 * 64 * 256 + 64 * 64 + 8 * 32 * 32) *
                sizeof(__bf16);
  lightning_attn_kernel<<<dim3(B_ * NH_), blk, smem, stream>>>(ws_qkv, ws_attn,
                                                               kv_final);
  // 4) attn = sigmoid_gate * rmsnorm(attn) (in place)
  rmsnorm_gate_kernel<<<dim3(M_), blk, 0, stream>>>(ws_attn, ws_gate, norm_w);
  // 5) out = attn @ out_w^T
  gemm_act_kernel<0><<<dim3(H_ / 128, M_ / 128), blk, 0, stream>>>(
      ws_attn, out_w, out, M_, H_, H_);
}
